// IDTBlock_32830730010669
// MI455X (gfx1250) — compile-verified, matrix operand order checked by reading, not run on hardware
//
#include <hip/hip_runtime.h>
#include <hip/hip_bf16.h>

typedef __bf16 bf16;
typedef __attribute__((ext_vector_type(16))) __bf16 bf16x16;
typedef __attribute__((ext_vector_type(8)))  float   f32x8;

#define HEADS 4
#define WIN 8
#define NW 64
#define CCH 64
#define HID 256
#define BDIM 4
#define HDIM 256
#define LTOK (HDIM*HDIM)
#define BW (BDIM*(HDIM/WIN)*(HDIM/WIN))   // 4096 windows
#define MTOK (BW*NW)                      // 262144 token rows
#define QKVN 256                          // packed q|k|v|vr columns

// ---- WMMA helpers ---------------------------------------------------------

__device__ __forceinline__ f32x8 wmma_bf16(bf16x16 a, bf16x16 b, f32x8 c) {
  return __builtin_amdgcn_wmma_f32_16x16x32_bf16(false, a, false, b, (short)0, c,
                                                 false, false);
}

// Load a 16x32 bf16 A-style fragment from LDS (row-major, row stride `ld`).
// Lane l holds row (row0 + l%16); K pattern per CDNA5 16-bit A layout:
// lanes 0-15: K = 0..7,16..23 ; lanes 16-31: K = 8..15,24..31 (relative k0).
// Also used for B by passing a transposed [N][K] panel.
__device__ __forceinline__ bf16x16 ldsfrag(const bf16* s, int ld, int row0,
                                           int k0, int lane) {
  const bf16* p = s + (size_t)(row0 + (lane & 15)) * ld + k0 + ((lane >> 4) << 3);
  bf16x16 f;
#pragma unroll
  for (int e = 0; e < 8; ++e) f[e] = p[e];
#pragma unroll
  for (int e = 0; e < 8; ++e) f[8 + e] = p[16 + e];
  return f;
}

__device__ __forceinline__ float gelu_exact(float v) {
  return 0.5f * v * (1.f + erff(v * 0.70710678118f));
}

// ---- Generic bf16 WMMA GEMM: C[M,N] = A[M,K] @ B[K,N] (+bias, epilogue) ----
// MODE 0: out bf16 = (acc+bias) * (n<64 ? scale : 1)   [fused qkv]
// MODE 1: out bf16 = gelu(acc+bias)
// MODE 2: out f32  = acc+bias+resid
// MODE 3: out f32  = acc+bias
template <int NT, int MODE>
__global__ __launch_bounds__(256) void gemm_ws(const bf16* __restrict__ A,
                                               const bf16* __restrict__ Bm,
                                               const float* __restrict__ bias,
                                               const float* __restrict__ resid,
                                               void* __restrict__ Out,
                                               int K, float scale) {
  constexpr int N = NT * 16;
  __shared__ bf16 sA[128 * 64];     // 16 KB A slab (128 rows x 64 K)
  __shared__ bf16 sBt[N * 64];      // transposed B chunk [N][64K], <=32 KB
  const int tid  = threadIdx.x;
  const int lane = tid & 31;
  const int wave = tid >> 5;
  const int m0   = blockIdx.x * 128;

  f32x8 acc[NT];                    // native WMMA accumulators (no mov traffic)
#pragma unroll
  for (int nt = 0; nt < NT; ++nt)
#pragma unroll
    for (int r = 0; r < 8; ++r) acc[nt][r] = 0.f;

  for (int kc = 0; kc < K; kc += 64) {
    // cooperative loads (uniform trip counts -> EXEC stays all-ones)
    for (int i = tid; i < 128 * 16; i += 256) {      // 4 bf16 per step
      int e = i * 4, r = e >> 6, c = e & 63;
      const bf16* g = A + (size_t)(m0 + r) * K + kc + c;
#pragma unroll
      for (int j = 0; j < 4; ++j) sA[r * 64 + c + j] = g[j];
    }
    if (kc + 64 < K) {               // hint next A chunk (global_prefetch_b8)
      int r = tid >> 1, c = (tid & 1) << 5;
      __builtin_prefetch(A + (size_t)(m0 + r) * K + kc + 64 + c, 0, 0);
    }
    for (int i = tid; i < N * 64; i += 256) {        // transpose B
      int n = i >> 6, k = i & 63;
      sBt[n * 64 + k] = Bm[(size_t)(kc + k) * N + n];
    }
    __syncthreads();
#pragma unroll
    for (int kt = 0; kt < 2; ++kt) {
      bf16x16 a = ldsfrag(sA, 64, wave * 16, kt * 32, lane);
#pragma unroll
      for (int nt = 0; nt < NT; ++nt) {
        bf16x16 b = ldsfrag(sBt, 64, nt * 16, kt * 32, lane);
        acc[nt] = wmma_bf16(a, b, acc[nt]);
      }
    }
    __syncthreads();
  }
  // epilogue: C layout -> VGPR r holds rows r / r+8, N = lane&15
  const int mbase = m0 + wave * 16 + ((lane >> 4) << 3);
  const int nc    = lane & 15;
#pragma unroll
  for (int nt = 0; nt < NT; ++nt) {
    int n = nt * 16 + nc;
    float bv = bias ? bias[n] : 0.f;
    float colscale = (MODE == 0) ? ((n < 64) ? scale : 1.f) : 1.f;
#pragma unroll
    for (int r = 0; r < 8; ++r) {
      size_t oidx = (size_t)(mbase + r) * N + n;
      float v = acc[nt][r] + bv;
      if (MODE == 0)      ((bf16*)Out)[oidx]  = (bf16)(v * colscale);
      else if (MODE == 1) ((bf16*)Out)[oidx]  = (bf16)gelu_exact(v);
      else if (MODE == 2) ((float*)Out)[oidx] = v + resid[oidx];
      else                ((float*)Out)[oidx] = v;
    }
  }
}

// ---- LN1 + window partition -> bf16 ---------------------------------------
__global__ void ln1_winpart(const float* __restrict__ x, const float* __restrict__ g,
                            const float* __restrict__ b, bf16* __restrict__ xw) {
  int t = blockIdx.x * blockDim.x + threadIdx.x;   // token id < MTOK
  int bi = t / LTOK, l = t % LTOK;
  int y = l >> 8, xx = l & 255;
  const float* px = x + (size_t)t * CCH;
  float v[CCH], mean = 0.f;
#pragma unroll
  for (int c = 0; c < CCH; ++c) { v[c] = px[c]; mean += v[c]; }
  mean *= (1.f / CCH);
  float var = 0.f;
#pragma unroll
  for (int c = 0; c < CCH; ++c) { float d = v[c] - mean; var += d * d; }
  var *= (1.f / CCH);
  float rs = rsqrtf(var + 1e-5f);
  int win = (bi * 32 + (y >> 3)) * 32 + (xx >> 3);
  int n   = ((y & 7) << 3) + (xx & 7);
  bf16* po = xw + ((size_t)win * NW + n) * CCH;
#pragma unroll
  for (int c = 0; c < CCH; ++c) po[c] = (bf16)((v[c] - mean) * rs * g[c] + b[c]);
}

// ---- Window attention: 1 block / window, 1 wave / head --------------------
// qkv packed [MTOK, 256]: q@0 (pre-scaled), k@64, v@128, v_r@192.
__global__ __launch_bounds__(128) void attn_win(const bf16* __restrict__ qkv,
                                                const float* __restrict__ rpb,
                                                float* __restrict__ oattn) {
  __shared__ float srpb[225 * HEADS];
  __shared__ bf16  sq [HEADS][64 * 32];   // q head slice, K padded 16->32
  __shared__ bf16  sk [HEADS][64 * 32];   // k head slice (acts as B^T)
  __shared__ bf16  svt[HEADS][16 * 64];   // v^T [d][kRow]
  __shared__ bf16  sp [HEADS][16 * 64];   // probability stripe
  const int tid = threadIdx.x, lane = tid & 31, h = tid >> 5;
  const int win = blockIdx.x;

  for (int i = tid; i < 225 * HEADS; i += 128) srpb[i] = rpb[i];
  const bf16* qg = qkv + (size_t)win * NW * QKVN + h * 16;         // q
  const bf16* kg = qg + 64;                                        // k
  const bf16* vg = qg + 128;                                       // v
  for (int i = lane; i < 64 * 32; i += 32) {
    int rr = i >> 5, cc = i & 31;
    sq[h][i] = (cc < 16) ? qg[(size_t)rr * QKVN + cc] : (bf16)0.f;
    sk[h][i] = (cc < 16) ? kg[(size_t)rr * QKVN + cc] : (bf16)0.f;
  }
  for (int i = lane; i < 16 * 64; i += 32) {
    int n = i >> 6, k = i & 63;
    svt[h][i] = vg[(size_t)k * QKVN + n];
  }
  __syncthreads();

  const int half8 = (lane >> 4) << 3;
  const int nc    = lane & 15;
  for (int mt = 0; mt < 4; ++mt) {
    bf16x16 a = ldsfrag(sq[h], 32, mt * 16, 0, lane);
    f32x8 sc[4];
#pragma unroll
    for (int nt = 0; nt < 4; ++nt) {
      f32x8 cz;
#pragma unroll
      for (int r = 0; r < 8; ++r) cz[r] = 0.f;
      sc[nt] = wmma_bf16(a, ldsfrag(sk[h], 32, nt * 16, 0, lane), cz);
    }
    // relative-position bias + row softmax (rows live in 16-lane halves)
#pragma unroll
    for (int r = 0; r < 8; ++r) {
      int m = mt * 16 + r + half8;
      int iyq = m >> 3, ixq = m & 7;
      float mx = -1e30f;
#pragma unroll
      for (int nt = 0; nt < 4; ++nt) {
        int n = nt * 16 + nc;
        int dy = iyq - (n >> 3) + 7, dx = ixq - (n & 7) + 7;
        sc[nt][r] += srpb[(dy * 15 + dx) * HEADS + h];
        mx = fmaxf(mx, sc[nt][r]);
      }
#pragma unroll
      for (int o = 1; o < 16; o <<= 1) mx = fmaxf(mx, __shfl_xor(mx, o, 32));
      float s = 0.f;
#pragma unroll
      for (int nt = 0; nt < 4; ++nt) { sc[nt][r] = __expf(sc[nt][r] - mx); s += sc[nt][r]; }
#pragma unroll
      for (int o = 1; o < 16; o <<= 1) s += __shfl_xor(s, o, 32);
      float inv = 1.f / s;
#pragma unroll
      for (int nt = 0; nt < 4; ++nt) sc[nt][r] *= inv;
    }
    // C-layout -> A-layout via LDS round trip (DS ops are in-order per wave)
#pragma unroll
    for (int nt = 0; nt < 4; ++nt)
#pragma unroll
      for (int r = 0; r < 8; ++r)
        sp[h][(r + half8) * 64 + nt * 16 + nc] = (bf16)sc[nt][r];
    f32x8 o;
#pragma unroll
    for (int r = 0; r < 8; ++r) o[r] = 0.f;
    o = wmma_bf16(ldsfrag(sp[h], 64, 0, 0,  lane), ldsfrag(svt[h], 64, 0, 0,  lane), o);
    o = wmma_bf16(ldsfrag(sp[h], 64, 0, 32, lane), ldsfrag(svt[h], 64, 0, 32, lane), o);
#pragma unroll
    for (int r = 0; r < 8; ++r) {
      int m = mt * 16 + r + half8;
      oattn[((size_t)win * NW + m) * CCH + h * 16 + nc] = o[r];
    }
  }
}

// ---- LePE position branch: dw5x5 + grouped 1x1, fused add -> bf16 ---------
__global__ __launch_bounds__(256) void pos_fuse(const bf16* __restrict__ qkv,
    const float* __restrict__ dww, const float* __restrict__ dwb,
    const float* __restrict__ pww, const float* __restrict__ pwb,
    const float* __restrict__ oattn, bf16* __restrict__ pin) {
  __shared__ float svr[NW * CCH];
  __shared__ float spe[NW * CCH];
  const int tid = threadIdx.x;
  const int win = blockIdx.x;
  const bf16* vrg = qkv + (size_t)win * NW * QKVN + 192;   // v_r slice
  for (int i = tid; i < NW * CCH; i += 256) {
    int n = i >> 6, c = i & 63;
    svr[n * CCH + c] = (float)vrg[(size_t)n * QKVN + c];
  }
  __syncthreads();
  for (int i = tid; i < NW * CCH; i += 256) {
    int n = i >> 6, c = i & 63;
    int iy = n >> 3, ix = n & 7;
    float s = dwb[c];
#pragma unroll
    for (int dy = 0; dy < 5; ++dy) {
      int yy = iy + dy - 2;
      if (yy < 0 || yy > 7) continue;
#pragma unroll
      for (int dx = 0; dx < 5; ++dx) {
        int xp = ix + dx - 2;
        if (xp < 0 || xp > 7) continue;
        s += svr[(yy * 8 + xp) * CCH + c] * dww[c * 25 + dy * 5 + dx];
      }
    }
    spe[n * CCH + c] = s;
  }
  __syncthreads();
  for (int i = tid; i < NW * CCH; i += 256) {
    int n = i >> 6, co = i & 63;
    int g = co >> 4;
    float s = pwb[co];
#pragma unroll
    for (int d = 0; d < 16; ++d) s += spe[n * CCH + g * 16 + d] * pww[co * 16 + d];
    size_t idx = ((size_t)win * NW + n) * CCH + co;
    pin[idx] = (bf16)(s + oattn[idx]);
  }
}

// ---- window reverse + residual + LN2 --------------------------------------
__global__ void winrev_res_ln2(const float* __restrict__ x, const float* __restrict__ aw,
                               const float* __restrict__ g, const float* __restrict__ bb,
                               float* __restrict__ x1, bf16* __restrict__ zin) {
  int t = blockIdx.x * blockDim.x + threadIdx.x;
  int bi = t / LTOK, l = t % LTOK;
  int y = l >> 8, xx = l & 255;
  int win = (bi * 32 + (y >> 3)) * 32 + (xx >> 3);
  int n   = ((y & 7) << 3) + (xx & 7);
  const float* pa = aw + ((size_t)win * NW + n) * CCH;
  const float* px = x + (size_t)t * CCH;
  float v[CCH], mean = 0.f;
#pragma unroll
  for (int c = 0; c < CCH; ++c) { v[c] = px[c] + pa[c]; mean += v[c]; }
  mean *= (1.f / CCH);
  float* po = x1 + (size_t)t * CCH;
  float var = 0.f;
#pragma unroll
  for (int c = 0; c < CCH; ++c) { po[c] = v[c]; float d = v[c] - mean; var += d * d; }
  var *= (1.f / CCH);
  float rs = rsqrtf(var + 1e-5f);
  bf16* pz = zin + (size_t)t * CCH;
#pragma unroll
  for (int c = 0; c < CCH; ++c) pz[c] = (bf16)((v[c] - mean) * rs * g[c] + bb[c]);
}

// ---- LeFF depthwise 3x3: z2 = gelu(conv(z)) + z ---------------------------
__global__ void leff_dw3(const bf16* __restrict__ z, const float* __restrict__ w,
                         const float* __restrict__ bb, bf16* __restrict__ z2) {
  size_t i = (size_t)blockIdx.x * blockDim.x + threadIdx.x;  // over B*L*HID
  int c = (int)(i & 255);
  size_t t = i >> 8;
  int l  = (int)(t & (LTOK - 1));
  int bi = (int)(t >> 16);
  int y = l >> 8, xx = l & 255;
  float s = bb[c];
#pragma unroll
  for (int dy = 0; dy < 3; ++dy) {
    int yy = y + dy - 1;
    if (yy < 0 || yy >= HDIM) continue;
#pragma unroll
    for (int dx = 0; dx < 3; ++dx) {
      int xp = xx + dx - 1;
      if (xp < 0 || xp >= HDIM) continue;
      s += (float)z[(((size_t)bi * LTOK) + (size_t)yy * HDIM + xp) * HID + c] *
           w[c * 9 + dy * 3 + dx];
    }
  }
  z2[i] = (bf16)(gelu_exact(s) + (float)z[i]);
}

// ---- weight packing / conversion ------------------------------------------
__global__ void f2bf(const float* __restrict__ in, bf16* __restrict__ out, int n) {
  int i = blockIdx.x * blockDim.x + threadIdx.x;
  if (i < n) out[i] = (bf16)in[i];
}

// pack [wq | wkv] -> [64,256] bf16 and [bq | bkv] -> [256] f32
__global__ void pack_qkv_w(const float* __restrict__ wq, const float* __restrict__ wkv,
                           const float* __restrict__ bq, const float* __restrict__ bkv,
                           bf16* __restrict__ w, float* __restrict__ bias) {
  int i = blockIdx.x * blockDim.x + threadIdx.x;   // 64*256
  if (i < 64 * QKVN) {
    int k = i >> 8, n = i & 255;
    float v = (n < 64) ? wq[k * 64 + n] : wkv[k * 192 + (n - 64)];
    w[i] = (bf16)v;
  }
  if (i < QKVN) bias[i] = (i < 64) ? bq[i] : bkv[i - 64];
}

// ---------------------------------------------------------------------------
extern "C" void kernel_launch(void* const* d_in, const int* in_sizes, int n_in,
                              void* d_out, int out_size, void* d_ws, size_t ws_size,
                              hipStream_t stream) {
  (void)in_sizes; (void)n_in; (void)out_size; (void)ws_size;
  const float* x       = (const float*)d_in[0];
  const float* n1g     = (const float*)d_in[1];
  const float* n1b     = (const float*)d_in[2];
  const float* wq      = (const float*)d_in[3];
  const float* bq      = (const float*)d_in[4];
  const float* wkv     = (const float*)d_in[5];
  const float* bkv     = (const float*)d_in[6];
  const float* rpb     = (const float*)d_in[7];
  const float* pe_dw_w = (const float*)d_in[8];
  const float* pe_dw_b = (const float*)d_in[9];
  const float* pe_pw_w = (const float*)d_in[10];
  const float* pe_pw_b = (const float*)d_in[11];
  const float* proj_w  = (const float*)d_in[12];
  const float* proj_b  = (const float*)d_in[13];
  const float* n2g     = (const float*)d_in[14];
  const float* n2b     = (const float*)d_in[15];
  const float* l1_w    = (const float*)d_in[16];
  const float* l1_b    = (const float*)d_in[17];
  const float* dw_w    = (const float*)d_in[18];
  const float* dw_b    = (const float*)d_in[19];
  const float* l2_w    = (const float*)d_in[20];
  const float* l2_b    = (const float*)d_in[21];
  float* out = (float*)d_out;

  char* base = (char*)d_ws;
  size_t off = 0;
  auto alloc = [&](size_t bytes) {
    size_t r = off;
    off = (off + bytes + 255) & ~(size_t)255;
    return r;
  };
  bf16*  xw_bf   = (bf16*)(base + alloc((size_t)MTOK * CCH * 2));
  bf16*  qkv_bf  = (bf16*)(base + alloc((size_t)MTOK * QKVN * 2));
  bf16*  wqkv_bf = (bf16*)(base + alloc(64 * QKVN * 2));
  float* bqkv    = (float*)(base + alloc(QKVN * 4));
  bf16*  pw_bf   = (bf16*)(base + alloc(64 * 64 * 2));
  bf16*  l1w_bf  = (bf16*)(base + alloc(64 * 256 * 2));
  bf16*  l2w_bf  = (bf16*)(base + alloc(256 * 64 * 2));
  float* oattn   = (float*)(base + alloc((size_t)MTOK * CCH * 4));
  bf16*  pin     = (bf16*)(base + alloc((size_t)MTOK * CCH * 2));
  float* aw      = (float*)(base + alloc((size_t)MTOK * CCH * 4));
  float* x1      = (float*)(base + alloc((size_t)MTOK * CCH * 4));
  bf16*  zin     = (bf16*)(base + alloc((size_t)MTOK * CCH * 2));
  bf16*  zb      = (bf16*)(base + alloc((size_t)MTOK * HID * 2));
  bf16*  z2b     = (bf16*)(base + alloc((size_t)MTOK * HID * 2));

  // weight packing / conversions
  pack_qkv_w<<<64, 256, 0, stream>>>(wq, wkv, bq, bkv, wqkv_bf, bqkv);
  f2bf<<<16, 256, 0, stream>>>(proj_w, pw_bf, 64 * 64);
  f2bf<<<64, 256, 0, stream>>>(l1_w, l1w_bf, 64 * 256);
  f2bf<<<64, 256, 0, stream>>>(l2_w, l2w_bf, 256 * 64);

  // 1) LN1 + window partition
  ln1_winpart<<<MTOK / 256, 256, 0, stream>>>(x, n1g, n1b, xw_bf);
  // 2) fused qkv: [q*hd^-0.5 | k | v | v_r]  (single pass over xw, N=256)
  gemm_ws<16, 0><<<MTOK / 128, 256, 0, stream>>>(xw_bf, wqkv_bf, bqkv, nullptr,
                                                 qkv_bf, 64, 0.25f);
  // 3) window attention (softmax + rel-pos bias) via WMMA
  attn_win<<<BW, 128, 0, stream>>>(qkv_bf, rpb, oattn);
  // 4) LePE branch + add
  pos_fuse<<<BW, 256, 0, stream>>>(qkv_bf, pe_dw_w, pe_dw_b, pe_pw_w, pe_pw_b,
                                   oattn, pin);
  // 5) proj (fp32 out, window layout)
  gemm_ws<4, 3><<<MTOK / 128, 256, 0, stream>>>(pin, pw_bf, proj_b, nullptr, aw,
                                                64, 1.f);
  // 6) window reverse + residual + LN2
  winrev_res_ln2<<<MTOK / 256, 256, 0, stream>>>(x, aw, n2g, n2b, x1, zin);
  // 7) LeFF l1 + GELU               [N=256]
  gemm_ws<16, 1><<<MTOK / 128, 256, 0, stream>>>(zin, l1w_bf, l1_b, nullptr, zb,
                                                 64, 1.f);
  // 8) depthwise 3x3 + GELU + residual
  leff_dw3<<<(size_t)MTOK * HID / 256, 256, 0, stream>>>(zb, dw_w, dw_b, z2b);
  // 9) LeFF l2 + residual -> final fp32 output
  gemm_ws<4, 2><<<MTOK / 128, 256, 0, stream>>>(z2b, l2w_bf, l2_b, x1, out,
                                                256, 1.f);
}
